// CAMLayer_17540646437466
// MI455X (gfx1250) — matrix operand order repaired and checked by reading, not compiled, not run on hardware
//
#include <hip/hip_runtime.h>

// Fused CAM layer for MI455X (gfx1250, wave32, WMMA):
//   y = ELU( concat(cam @ x3, x3) @ W.T + b )
//
// All matrix math on the WMMA pipes:
//   stage A: x_agg.T = x3.T @ cam.T per batch -> K=21 (pad 32) => 1 wmma per
//            16x16 tile, 192 wmma/block; D lane->joint makes the validity
//            guard lane-uniform and the write-back a single ds_store_b128.
//   stage B: y = x_cat(128x512) @ W.T chunks(512x64) -> 2048 wmma/block.
//
// HBM traffic ~0.7 GB (x in + y out) -> ~30us at 23.3 TB/s; bf16 WMMA compute
// is comparable, so everything (agg, concat, GEMM, bias, ELU) is fused into
// one pass with x staged once in LDS.

#define NJ      21
#define FIN     256
#define KCAT    512
#define NOUT    256
#define TB      6
#define ROWS_PB (TB * NJ)        // 126 valid rows per block
#define MT      128              // padded M tile
#define XSTR    520              // sX row stride in halves (512 + 8 -> bank step 4 dw)
#define XTSTR   200              // sXT row stride in halves (6*32 + 8)
#define CSTR    40               // sCamB row stride in halves (32 + 8)
#define NCHUNK  64               // output columns per W chunk (4 chunks)

#define SX_H   (MT * XSTR)       // 66,560 halves
#define SXT_H  (FIN * XTSTR)     // 51,200 halves
#define SW_H   (NCHUNK * XSTR)   // 33,280 halves
#define SCAM_H (32 * CSTR)       // 1,280 halves
#define SMEM_BYTES ((SX_H + SXT_H + SW_H + SCAM_H) * 2 + NOUT * 4)   // 305,664

typedef __attribute__((ext_vector_type(16))) __bf16       v16bf;
typedef __attribute__((ext_vector_type(8)))  float        v8f;
typedef __attribute__((ext_vector_type(4)))  unsigned int v4u;

union FragU { v4u v[2]; v16bf f; };

__device__ __forceinline__ unsigned int f2bf(float f) {
  unsigned int u = __float_as_uint(f);
  u += 0x7FFFu + ((u >> 16) & 1u);   // round-to-nearest-even
  return u >> 16;
}

// A fragment, 16x32 bf16 (ISA 7.12.2): lane L holds row M = L&15;
// halves 0..7 = K[hs*8 .. +7], halves 8..15 = K[16+hs*8 .. +7], hs = L>>4.
__device__ __forceinline__ v16bf ldsFragA(const unsigned short* s, int stride,
                                          int row0, int k0, int lane) {
  const unsigned short* p = s + (row0 + (lane & 15)) * stride + k0 + ((lane >> 4) << 3);
  FragU u;
  u.v[0] = *(const v4u*)p;
  u.v[1] = *(const v4u*)(p + 16);
  return u.f;
}

// B fragment, 32x16 bf16: lane L holds column N = L&15; halves 0..7 =
// K[hs*8 .. +7], halves 8..15 = K[16+hs*8 .. +7]. Source stored row-major [n][k].
__device__ __forceinline__ v16bf ldsFragB(const unsigned short* s, int stride,
                                          int col0, int k0, int lane) {
  const unsigned short* p = s + (col0 + (lane & 15)) * stride + k0 + ((lane >> 4) << 3);
  FragU u;
  u.v[0] = *(const v4u*)p;
  u.v[1] = *(const v4u*)(p + 16);
  return u.f;
}

__device__ __forceinline__ unsigned long long pack4bf(float a, float b, float c, float d) {
  return (unsigned long long)f2bf(a)         |
         ((unsigned long long)f2bf(b) << 16) |
         ((unsigned long long)f2bf(c) << 32) |
         ((unsigned long long)f2bf(d) << 48);
}

__global__ __launch_bounds__(256, 1)
void cam_fused_wmma(const float* __restrict__ x,
                    const float* __restrict__ cam,
                    const float* __restrict__ W,
                    const float* __restrict__ bias,
                    float* __restrict__ out,
                    int total_rows) {
  extern __shared__ unsigned char smem_raw[];
  unsigned short* sX    = (unsigned short*)smem_raw;   // [MT][XSTR]   x_cat tile
  unsigned short* sXT   = sX + SX_H;                   // [FIN][XTSTR] x transposed (agg A)
  unsigned short* sW    = sXT + SXT_H;                 // [NCHUNK][XSTR] W chunk
  unsigned short* sCamB = sW + SW_H;                   // [32][CSTR]   cam bf16 [i][j] (agg B)
  float*          sBias = (float*)(sCamB + SCAM_H);    // [NOUT]

  const int tid  = threadIdx.x;
  const int lane = tid & 31;
  const int wid  = tid >> 5;
  const int hs   = lane >> 4;
  const int ln   = lane & 15;

  const int row0g      = blockIdx.x * ROWS_PB;
  const int rows_valid = min(ROWS_PB, total_rows - row0g);

  // ---- stage 0: zero the zero-padded operands; bias to LDS ---------------
  {
    unsigned long long* z = (unsigned long long*)sXT;
    for (int e = tid; e < SXT_H / 4; e += 256) z[e] = 0ull;
    unsigned long long* zc = (unsigned long long*)sCamB;
    for (int e = tid; e < SCAM_H / 4; e += 256) zc[e] = 0ull;
    for (int e = tid; e < NOUT; e += 256) sBias[e] = bias[e];
  }
  __syncthreads();

  // ---- stage 1: cam -> bf16 [i][j]; x -> sX right half + sXT transposed --
  for (int e = tid; e < NJ * NJ; e += 256) {
    const int i = e / NJ, j = e - i * NJ;
    sCamB[i * CSTR + j] = (unsigned short)f2bf(cam[e]);
  }
  {
    const float4* x4 = (const float4*)(x + (size_t)row0g * FIN);
    for (int e = tid; e < ROWS_PB * (FIN / 4); e += 256) {
      const int r  = e >> 6;                     // FIN/4 = 64 float4 per row
      const int kg = e & 63;
      if (r < rows_valid) {
        const float4 v = x4[r * (FIN / 4) + kg];
        // row-major copy (x_cat right half, GEMM A operand)
        *(unsigned long long*)(sX + r * XSTR + FIN + kg * 4) =
            pack4bf(v.x, v.y, v.z, v.w);
        // transposed copy (agg A operand): sXT[f][t*32 + j]
        const int t  = r / NJ;
        const int jj = r - t * NJ;
        const int cc = t * 32 + jj;
        const int f0 = kg * 4;
        sXT[(f0 + 0) * XTSTR + cc] = (unsigned short)f2bf(v.x);
        sXT[(f0 + 1) * XTSTR + cc] = (unsigned short)f2bf(v.y);
        sXT[(f0 + 2) * XTSTR + cc] = (unsigned short)f2bf(v.z);
        sXT[(f0 + 3) * XTSTR + cc] = (unsigned short)f2bf(v.w);
      }
    }
  }
  __syncthreads();

  // ---- stage 2: aggregation on the WMMA pipes (transposed) ---------------
  // x_agg.T(t)[f, i] = sum_j x.T(t)[f, j] * cam[i, j]
  //   A = sXT rows f (16), K = j (32, zero-padded)
  //   B = sCamB viewed as [n=i][k=j] row-major -> ldsFragB directly
  //   D: lane -> i (guard lane-uniform), vgpr r -> f (8 consecutive features)
  // 6 t * 16 ft * 2 ni = 192 tiles, round-robin over the 8 waves.
  {
    const v8f zero = (v8f)0.0f;
    for (int idx = wid; idx < TB * (FIN / 16) * 2; idx += 8) {
      const int ni = idx & 1;             // joint-column tile (N)
      const int ft = (idx >> 1) & 15;     // feature tile (M)
      const int t  = idx >> 5;            // batch in block
      v16bf a = ldsFragA(sXT, XTSTR, ft * 16, t * 32, lane);
      v16bf b = ldsFragB(sCamB, CSTR, ni * 16, 0, lane);
      v8f d = __builtin_amdgcn_wmma_f32_16x16x32_bf16(
          false, a, false, b, (short)0, zero, false, false);
      const int iv = ni * 16 + ln;        // joint index (lane-uniform guard)
      if (iv < NJ) {
        const v4u pk = { f2bf(d[0]) | (f2bf(d[1]) << 16),
                         f2bf(d[2]) | (f2bf(d[3]) << 16),
                         f2bf(d[4]) | (f2bf(d[5]) << 16),
                         f2bf(d[6]) | (f2bf(d[7]) << 16) };
        *(v4u*)(sX + (t * NJ + iv) * XSTR + ft * 16 + hs * 8) = pk;
      }
    }
  }
  __syncthreads();

  // ---- stage 3: main GEMM, four 64-col output chunks ---------------------
  const int mg = wid >> 1;        // wave rows: mg*32 .. +31
  const int ng = wid & 1;         // wave cols within chunk: ng*32 .. +31

  for (int chunk = 0; chunk < NOUT / NCHUNK; ++chunk) {
    // load W chunk rows [chunk*64, +64), fp32 -> bf16, row-major [n][k]
    const float4* w4 = (const float4*)(W + (size_t)chunk * NCHUNK * KCAT);
    for (int e = tid; e < NCHUNK * (KCAT / 4); e += 256) {
      const int n  = e >> 7;                   // KCAT/4 = 128 float4 per row
      const int kg = e & 127;
      const float4 v = w4[n * (KCAT / 4) + kg];
      *(unsigned long long*)(sW + n * XSTR + kg * 4) = pack4bf(v.x, v.y, v.z, v.w);
    }
    // prefetch next chunk of W into cache while this chunk computes
    if (chunk + 1 < NOUT / NCHUNK) {
      const char* nw = (const char*)(W + (size_t)(chunk + 1) * NCHUNK * KCAT);
      for (int e = tid; e < (NCHUNK * KCAT * 4) / 256; e += 256)
        __builtin_prefetch(nw + e * 256, 0, 1);
    }
    __syncthreads();

    v8f acc[2][2];
#pragma unroll
    for (int mi = 0; mi < 2; ++mi)
#pragma unroll
      for (int ni = 0; ni < 2; ++ni)
        acc[mi][ni] = (v8f)0.0f;

    for (int ks = 0; ks < KCAT / 32; ++ks) {
      const int k0 = ks * 32;
      v16bf a[2], b[2];
#pragma unroll
      for (int mi = 0; mi < 2; ++mi)
        a[mi] = ldsFragA(sX, XSTR, mg * 32 + mi * 16, k0, lane);
#pragma unroll
      for (int ni = 0; ni < 2; ++ni)
        b[ni] = ldsFragB(sW, XSTR, ng * 32 + ni * 16, k0, lane);
#pragma unroll
      for (int mi = 0; mi < 2; ++mi)
#pragma unroll
        for (int ni = 0; ni < 2; ++ni)
          acc[mi][ni] = __builtin_amdgcn_wmma_f32_16x16x32_bf16(
              false, a[mi], false, b[ni], (short)0, acc[mi][ni], false, false);
    }

    // ---- epilogue: bias + ELU, guarded store (C/D layout: lane->N, vgpr->M)
#pragma unroll
    for (int mi = 0; mi < 2; ++mi) {
#pragma unroll
      for (int ni = 0; ni < 2; ++ni) {
        const int col = chunk * NCHUNK + ng * 32 + ni * 16 + ln;
        const float bv = sBias[col];
#pragma unroll
        for (int r = 0; r < 8; ++r) {
          const int rl = mg * 32 + mi * 16 + hs * 8 + r;
          if (rl < rows_valid) {
            float y = acc[mi][ni][r] + bv;
            y = (y > 0.f) ? y : (__expf(y) - 1.f);
            out[(size_t)(row0g + rl) * NOUT + col] = y;
          }
        }
      }
    }
    __syncthreads();   // before next chunk overwrites sW
  }
}

extern "C" void kernel_launch(void* const* d_in, const int* in_sizes, int n_in,
                              void* d_out, int out_size, void* d_ws, size_t ws_size,
                              hipStream_t stream) {
  (void)n_in; (void)out_size; (void)d_ws; (void)ws_size;
  const float* x    = (const float*)d_in[0];
  const float* cam  = (const float*)d_in[1];
  const float* W    = (const float*)d_in[2];
  const float* bias = (const float*)d_in[3];
  float* out = (float*)d_out;

  const int total_rows = in_sizes[0] / FIN;                 // 344064
  const int grid = (total_rows + ROWS_PB - 1) / ROWS_PB;    // 2731

  cam_fused_wmma<<<dim3(grid), dim3(256), SMEM_BYTES, stream>>>(
      x, cam, W, bias, out, total_rows);
}
